// Qwen3VLTextAttention_50800873177176
// MI455X (gfx1250) — matrix operand order repaired and checked by reading, not compiled, not run on hardware
//
#include <hip/hip_runtime.h>
#include <hip/hip_bf16.h>

// ---------------------------------------------------------------------------
// Qwen3VL text attention block for gfx1250 (MI455X), bf16 WMMA path.
// B=2, S=2048, HID=2048, H=16, KV=4, D=128.
// ---------------------------------------------------------------------------

#define CB 2
#define CS 2048
#define CHID 2048
#define CH 16
#define CKV 4
#define CD 128
#define CM (CB * CS) // 4096 rows

typedef __attribute__((ext_vector_type(16))) __bf16 v16bf;
typedef __attribute__((ext_vector_type(8)))  float  v8f;
typedef __attribute__((ext_vector_type(4)))  float  f32x4;
typedef __attribute__((ext_vector_type(2)))  float  f32x2;

union FragU { v16bf v; f32x4 q[2]; };
union OutU  { f32x4 f; __bf16 h[8]; };
union Pack4 { __bf16 h[4]; f32x2 d; };

static __device__ __forceinline__ v8f wmma_bf16(v16bf a, v16bf b, v8f c) {
  return __builtin_amdgcn_wmma_f32_16x16x32_bf16(false, a, false, b, (short)0, c,
                                                 false, false);
}

// A-matrix fragment (16x32 bf16, row-major source, leading dim ld):
// lane L: row = L%16; lanes<16 hold K 0..15, lanes>=16 hold K 16..31.
static __device__ __forceinline__ v16bf load_afrag(const __bf16* __restrict__ base,
                                                   int ld) {
  int lane = threadIdx.x & 31;
  int row  = lane & 15;
  int kh   = lane & 16; // 0 for lanes 0-15, 16 for lanes 16-31
  const __bf16* p = base + (size_t)row * ld + kh;
  FragU u;
  u.q[0] = *(const f32x4*)p;        // K = kh .. kh+7
  u.q[1] = *(const f32x4*)(p + 8);  // K = kh+8 .. kh+15
  return u.v;
}

// B-matrix fragment (32x16 bf16) where B = R^T of row-major R (N x K, ld):
// lane L: col n = L%16; lanes<16 hold K 0..15, lanes>=16 hold K 16..31.
static __device__ __forceinline__ v16bf load_bfrag(const __bf16* __restrict__ base,
                                                   int ld) {
  int lane = threadIdx.x & 31;
  int row  = lane & 15;
  int koff = lane & 16; // 0 or 16
  const __bf16* p = base + (size_t)row * ld + koff;
  FragU u;
  u.q[0] = *(const f32x4*)p;       // K = koff .. koff+7
  u.q[1] = *(const f32x4*)(p + 8); // K = koff+8 .. koff+15
  return u.v;
}

// Shared 64x64-per-wave WMMA GEMM body: C(MxN,f32) = A @ W^T, K-reduction.
static __device__ __forceinline__ void gemm_tile_64x64(
    const __bf16* __restrict__ A, const __bf16* __restrict__ W,
    float* __restrict__ C, int N, int K, int mbase, int nbase) {
  int lane = threadIdx.x & 31;
  v8f acc[4][4] = {};
  for (int k = 0; k < K; k += 32) {
    v16bf a[4], b[4];
#pragma unroll
    for (int i = 0; i < 4; ++i)
      a[i] = load_afrag(A + (size_t)(mbase + 16 * i) * K + k, K);
#pragma unroll
    for (int j = 0; j < 4; ++j)
      b[j] = load_bfrag(W + (size_t)(nbase + 16 * j) * K + k, K);
#pragma unroll
    for (int i = 0; i < 4; ++i)
#pragma unroll
      for (int j = 0; j < 4; ++j)
        acc[i][j] = wmma_bf16(a[i], b[j], acc[i][j]);
  }
  int q  = lane & 15;
  int hu = lane >> 4;
#pragma unroll
  for (int i = 0; i < 4; ++i)
#pragma unroll
    for (int j = 0; j < 4; ++j)
#pragma unroll
      for (int g = 0; g < 8; ++g) {
        int m = mbase + 16 * i + g + 8 * hu;
        int n = nbase + 16 * j + q;
        C[(size_t)m * N + n] = acc[i][j][g];
      }
}

// ---------------------------------------------------------------------------
// Generic GEMM: one wave per 64x64 tile.
// ---------------------------------------------------------------------------
__global__ __launch_bounds__(32) void gemm_xwt(const __bf16* __restrict__ A,
                                               const __bf16* __restrict__ W,
                                               float* __restrict__ C,
                                               int N, int K) {
  gemm_tile_64x64(A, W, C, N, K, blockIdx.y * 64, blockIdx.x * 64);
}

// ---------------------------------------------------------------------------
// Fused QKV projection: one launch covering N = 2048 (Q) + 512 (K) + 512 (V).
// grid.x = 48 (32 Q tiles, 8 K tiles, 8 V tiles), grid.y = M/64.
// Keeps all ~1024 SIMDs busy instead of three serialized launches.
// ---------------------------------------------------------------------------
__global__ __launch_bounds__(32) void gemm_qkv(const __bf16* __restrict__ A,
                                               const __bf16* __restrict__ Wq,
                                               const __bf16* __restrict__ Wk,
                                               const __bf16* __restrict__ Wv,
                                               float* __restrict__ Qf,
                                               float* __restrict__ Kf,
                                               float* __restrict__ Vf) {
  int j = blockIdx.x;
  int mbase = blockIdx.y * 64;
  const __bf16* W;
  float* C;
  int N, nbase;
  if (j < 32) {        // Q: N = H*D = 2048
    W = Wq; C = Qf; N = CH * CD; nbase = j * 64;
  } else if (j < 40) { // K: N = KV*D = 512
    W = Wk; C = Kf; N = CKV * CD; nbase = (j - 32) * 64;
  } else {             // V: N = KV*D = 512
    W = Wv; C = Vf; N = CKV * CD; nbase = (j - 40) * 64;
  }
  gemm_tile_64x64(A, W, C, N, CHID, mbase, nbase);
}

// ---------------------------------------------------------------------------
// f32 -> bf16 elementwise (vectorized x4)
// ---------------------------------------------------------------------------
__global__ void f32_to_bf16(const float* __restrict__ in,
                            __bf16* __restrict__ out, int n4) {
  int i = blockIdx.x * blockDim.x + threadIdx.x;
  if (i >= n4) return;
  f32x4 v = ((const f32x4*)in)[i];
  Pack4 u;
  u.h[0] = (__bf16)v.x; u.h[1] = (__bf16)v.y;
  u.h[2] = (__bf16)v.z; u.h[3] = (__bf16)v.w;
  ((f32x2*)out)[i] = u.d;
}

// ---------------------------------------------------------------------------
// Fused RMSNorm (per D=128 head vector) + RoPE + scale, f32 -> bf16.
// One wave per row; rows ordered (b, s, head). nh = heads per token.
// ---------------------------------------------------------------------------
__global__ __launch_bounds__(128) void rmsnorm_rope(const float* __restrict__ X,
                                                    const float* __restrict__ cosb,
                                                    const float* __restrict__ sinb,
                                                    const float* __restrict__ w,
                                                    __bf16* __restrict__ out,
                                                    int nrows, int nh, float scale) {
  int row = blockIdx.x * 4 + (threadIdx.x >> 5);
  if (row >= nrows) return;
  int lane = threadIdx.x & 31;
  const float* x = X + (size_t)row * CD;
  f32x4 v = ((const f32x4*)x)[lane];
  float ss = v.x * v.x + v.y * v.y + v.z * v.z + v.w * v.w;
  ss += __shfl_xor(ss, 1, 32);
  ss += __shfl_xor(ss, 2, 32);
  ss += __shfl_xor(ss, 4, 32);
  ss += __shfl_xor(ss, 8, 32);
  ss += __shfl_xor(ss, 16, 32);
  float inv = rsqrtf(ss * (1.0f / (float)CD) + 1e-6f);
  f32x4 wv = ((const f32x4*)w)[lane];
  f32x4 xn = v * inv * wv;
  // RoPE partner: d and d^64 live in lanes L and L^16.
  f32x4 pr;
  pr.x = __shfl_xor(xn.x, 16, 32);
  pr.y = __shfl_xor(xn.y, 16, 32);
  pr.z = __shfl_xor(xn.z, 16, 32);
  pr.w = __shfl_xor(xn.w, 16, 32);
  float sgn = (lane < 16) ? -1.0f : 1.0f; // -x2 for d<64, +x1 for d>=64
  size_t cofs = (size_t)(row / nh) * CD + lane * 4;
  f32x4 cs = *(const f32x4*)(cosb + cofs);
  f32x4 sn = *(const f32x4*)(sinb + cofs);
  f32x4 o = (xn * cs + pr * sn * sgn) * scale;
  Pack4 u;
  u.h[0] = (__bf16)o.x; u.h[1] = (__bf16)o.y;
  u.h[2] = (__bf16)o.z; u.h[3] = (__bf16)o.w;
  *(f32x2*)(out + (size_t)row * CD + lane * 4) = u.d;
}

// ---------------------------------------------------------------------------
// V (B,S,KV,D) f32  ->  Vt (B,KV,D,S) bf16
// ---------------------------------------------------------------------------
__global__ void transpose_v(const float* __restrict__ Vf,
                            __bf16* __restrict__ Vt) {
  int idx = blockIdx.x * blockDim.x + threadIdx.x; // over B*KV*D*S
  int s  = idx & (CS - 1);
  int r  = idx >> 11;
  int d  = r & (CD - 1);
  r >>= 7;
  int kv = r & (CKV - 1);
  int b  = r >> 2;
  Vt[idx] = (__bf16)Vf[(((size_t)(b * CS + s)) * CKV + kv) * CD + d];
}

// ---------------------------------------------------------------------------
// Causal GQA flash attention. One wave per (b, h, 16-query tile).
// Computes S^T = K * Q^T (stats per-lane) and O^T = V^T * P^T.
// Q already scaled by 1/sqrt(D). Output AO bf16 (B,S,H*D).
// ---------------------------------------------------------------------------
__global__ __launch_bounds__(32) void attn_fwd(const __bf16* __restrict__ Qb,
                                               const __bf16* __restrict__ Kb,
                                               const __bf16* __restrict__ Vt,
                                               __bf16* __restrict__ AO) {
  int qbase = blockIdx.x * 16;
  int h     = blockIdx.y;
  int b     = blockIdx.z;
  int kvh   = h >> 2; // H/KV = 4 query heads per kv head
  int lane  = threadIdx.x & 31;
  int q     = lane & 15;
  int hu    = lane >> 4;
  int qi    = qbase + q;

  const __bf16* qrow = Qb + ((size_t)(b * CS + qbase) * CH + h) * CD;
  v16bf qf[4];
#pragma unroll
  for (int ds = 0; ds < 4; ++ds)
    qf[ds] = load_bfrag(qrow + ds * 32, CH * CD);

  const __bf16* kbh = Kb + (size_t)b * CS * CKV * CD + (size_t)kvh * CD;
  const __bf16* vtb = Vt + ((size_t)(b * CKV + kvh)) * CD * CS;

  v8f acc[8] = {};
  float mstat = -3.0e38f, lstat = 0.0f;

  for (int kvb = 0; kvb <= qbase + 15; kvb += 32) {
    // ---- scores: S^T tiles (rows = kv, cols = q) ----
    v8f st0 = {}, st1 = {};
#pragma unroll
    for (int ds = 0; ds < 4; ++ds) {
      v16bf k0 = load_afrag(kbh + (size_t)kvb * CKV * CD + ds * 32, CKV * CD);
      v16bf k1 = load_afrag(kbh + (size_t)(kvb + 16) * CKV * CD + ds * 32, CKV * CD);
      st0 = wmma_bf16(k0, qf[ds], st0);
      st1 = wmma_bf16(k1, qf[ds], st1);
    }
    // ---- causal mask ----
#pragma unroll
    for (int g = 0; g < 8; ++g) {
      int kv0 = kvb + g + 8 * hu;
      st0[g] = (kv0 <= qi) ? st0[g] : -3.0e38f;
      st1[g] = (kv0 + 16 <= qi) ? st1[g] : -3.0e38f;
    }
    // ---- online softmax over kv (per-lane column stats) ----
    float bm = -3.0e38f;
#pragma unroll
    for (int g = 0; g < 8; ++g) bm = fmaxf(bm, fmaxf(st0[g], st1[g]));
    bm = fmaxf(bm, __shfl_xor(bm, 16, 32));
    float nm   = fmaxf(mstat, bm);
    float corr = __expf(mstat - nm);
    float rs   = 0.0f;
#pragma unroll
    for (int g = 0; g < 8; ++g) {
      st0[g] = __expf(st0[g] - nm);
      st1[g] = __expf(st1[g] - nm);
      rs += st0[g] + st1[g];
    }
    rs += __shfl_xor(rs, 16, 32);
    lstat = lstat * corr + rs;
    mstat = nm;
#pragma unroll
    for (int dt = 0; dt < 8; ++dt) acc[dt] = acc[dt] * corr;

    // ---- build P^T B-fragment from S^T C-layout tiles ----
    float s0s[8], s1s[8];
#pragma unroll
    for (int g = 0; g < 8; ++g) {
      s0s[g] = __shfl_xor(st0[g], 16, 32);
      s1s[g] = __shfl_xor(st1[g], 16, 32);
    }
    bool lo = lane < 16;
    v16bf pb;
#pragma unroll
    for (int vv = 0; vv < 4; ++vv) { // K = koff+2v.. from tile[halfdst] rows 0..7
      float e0 = lo ? st0[2 * vv]     : s1s[2 * vv];
      float e1 = lo ? st0[2 * vv + 1] : s1s[2 * vv + 1];
      pb[2 * vv]     = (__bf16)e0;
      pb[2 * vv + 1] = (__bf16)e1;
    }
#pragma unroll
    for (int vv = 4; vv < 8; ++vv) { // rows 8..15 of selected tile
      float e0 = lo ? s0s[2 * vv - 8] : st1[2 * vv - 8];
      float e1 = lo ? s0s[2 * vv - 7] : st1[2 * vv - 7];
      pb[2 * vv]     = (__bf16)e0;
      pb[2 * vv + 1] = (__bf16)e1;
    }
    // ---- O^T += V^T * P^T ----
#pragma unroll
    for (int dt = 0; dt < 8; ++dt) {
      v16bf va = load_afrag(vtb + (size_t)(dt * 16) * CS + kvb, CS);
      acc[dt] = wmma_bf16(va, pb, acc[dt]);
    }
  }

  float invl = 1.0f / lstat;
  __bf16* orow = AO + ((size_t)(b * CS + qbase + q) * CH + h) * CD + 8 * hu;
#pragma unroll
  for (int dt = 0; dt < 8; ++dt) {
    OutU u;
#pragma unroll
    for (int g = 0; g < 8; ++g) u.h[g] = (__bf16)(acc[dt][g] * invl);
    *(f32x4*)(orow + dt * 16) = u.f; // 8 consecutive d, 16B store
  }
}

// ---------------------------------------------------------------------------
// Host-side orchestration
// ---------------------------------------------------------------------------
extern "C" void kernel_launch(void* const* d_in, const int* in_sizes, int n_in,
                              void* d_out, int out_size, void* d_ws, size_t ws_size,
                              hipStream_t stream) {
  const float* hidden = (const float*)d_in[0];
  const float* cosb   = (const float*)d_in[1];
  const float* sinb   = (const float*)d_in[2];
  const float* Wq     = (const float*)d_in[3];
  const float* Wk     = (const float*)d_in[4];
  const float* Wv     = (const float*)d_in[5];
  const float* Wo     = (const float*)d_in[6];
  const float* qnw    = (const float*)d_in[7];
  const float* knw    = (const float*)d_in[8];
  float* out = (float*)d_out;

  // workspace carve (256B aligned blocks)
  char* p = (char*)d_ws;
  auto carve = [&](size_t bytes) -> void* {
    void* r = (void*)p;
    p += (bytes + 255) & ~(size_t)255;
    return r;
  };
  const size_t nHid = (size_t)CM * CHID;       // 8388608
  const size_t nQ   = (size_t)CM * CH * CD;    // 8388608
  const size_t nK   = (size_t)CM * CKV * CD;   // 2097152
  __bf16* hb  = (__bf16*)carve(nHid * 2);
  __bf16* wqb = (__bf16*)carve((size_t)CH * CD * CHID * 2);
  __bf16* wkb = (__bf16*)carve((size_t)CKV * CD * CHID * 2);
  __bf16* wvb = (__bf16*)carve((size_t)CKV * CD * CHID * 2);
  __bf16* wob = (__bf16*)carve((size_t)CHID * CH * CD * 2);
  float*  Qf  = (float*)carve(nQ * 4);
  float*  Kf  = (float*)carve(nK * 4);
  float*  Vf  = (float*)carve(nK * 4);
  __bf16* Qbh = (__bf16*)carve(nQ * 2);
  __bf16* Kbh = (__bf16*)carve(nK * 2);
  __bf16* Vtb = (__bf16*)carve(nK * 2);
  __bf16* AOb = (__bf16*)carve(nQ * 2);

  // 1) f32 -> bf16 conversions
  auto cvt = [&](const float* src, __bf16* dst, size_t n) {
    int n4 = (int)(n / 4);
    f32_to_bf16<<<(n4 + 255) / 256, 256, 0, stream>>>(src, dst, n4);
  };
  cvt(hidden, hb, nHid);
  cvt(Wq, wqb, (size_t)CH * CD * CHID);
  cvt(Wk, wkb, (size_t)CKV * CD * CHID);
  cvt(Wv, wvb, (size_t)CKV * CD * CHID);
  cvt(Wo, wob, (size_t)CHID * CH * CD);

  // 2) fused QKV projection (single WMMA GEMM launch, 3072 waves)
  {
    dim3 g(48, CM / 64); // 32 Q-tiles + 8 K-tiles + 8 V-tiles in N
    gemm_qkv<<<g, 32, 0, stream>>>(hb, wqb, wkb, wvb, Qf, Kf, Vf);
  }

  // 3) RMSNorm + RoPE (fold 1/sqrt(D) into Q)
  {
    const float qscale = 0.08838834764831845f; // 1/sqrt(128)
    int qrows = CM * CH;
    int krows = CM * CKV;
    rmsnorm_rope<<<qrows / 4, 128, 0, stream>>>(Qf, cosb, sinb, qnw, Qbh, qrows,
                                                CH, qscale);
    rmsnorm_rope<<<krows / 4, 128, 0, stream>>>(Kf, cosb, sinb, knw, Kbh, krows,
                                                CKV, 1.0f);
  }

  // 4) V transpose-convert to (B,KV,D,S) bf16
  transpose_v<<<(int)(nK / 256), 256, 0, stream>>>(Vf, Vtb);

  // 5) causal flash attention
  {
    dim3 ga(CS / 16, CH, CB);
    attn_fwd<<<ga, 32, 0, stream>>>(Qbh, Kbh, Vtb, AOb);
  }

  // 6) output projection into d_out (f32)
  {
    dim3 go(CHID / 64, CM / 64);
    gemm_xwt<<<go, 32, 0, stream>>>(AOb, wob, out, CHID, CH * CD);
  }
}
